// GNN_47107201302528
// MI455X (gfx1250) — matrix operand order repaired and checked by reading, compiled, not run on hardware
//
#include <hip/hip_runtime.h>

#define NNODES 50000
#define NEDGES 800000
#define DH 64

typedef _Float16 v16h __attribute__((ext_vector_type(16)));
typedef _Float16 v8h  __attribute__((ext_vector_type(8)));
typedef _Float16 h4   __attribute__((ext_vector_type(4)));
typedef float    v8f  __attribute__((ext_vector_type(8)));
typedef int      v4i  __attribute__((vector_size(4 * sizeof(int))));

// Async global->LDS path (CDNA5): guarded so the file compiles on any toolchain.
#if defined(__has_builtin)
#  if __has_builtin(__builtin_amdgcn_global_load_async_to_lds_b128)
#    define HAVE_ASYNC_LDS 1
#  endif
#endif
#ifndef HAVE_ASYNC_LDS
#  define HAVE_ASYNC_LDS 0
#endif

// ---------------------------------------------------------------- helpers

__device__ __forceinline__ float gelu_f(float x) {
  const float c = 0.7978845608028654f; // sqrt(2/pi)
  float t = tanhf(c * (x + 0.044715f * x * x * x));
  return 0.5f * x * (1.0f + t);
}

// Intra-wave LDS producer->consumer ordering (cross-lane via LDS):
// DS ops of one wave complete in order; wait the DS counter and fence the scheduler.
__device__ __forceinline__ void wave_lds_sync() {
  __builtin_amdgcn_wave_barrier();
  asm volatile("s_wait_dscnt 0" ::: "memory");
  __builtin_amdgcn_wave_barrier();
}

__device__ __forceinline__ void wait_async_all() {
#if HAVE_ASYNC_LDS
#  if __has_builtin(__builtin_amdgcn_s_wait_asynccnt)
  __builtin_amdgcn_s_wait_asynccnt(0);
#  else
  asm volatile("s_wait_asynccnt 0" ::: "memory");
#  endif
#endif
}

// Cooperative (block-wide) staging of packed f16 weights into LDS,
// 16 bytes per lane per iteration. Uses the ASYNCcnt-tracked direct
// global->LDS copy when available.
__device__ __forceinline__ void stage_weights(_Float16* dst, const _Float16* src,
                                              int halves, int t) {
  int n16 = halves >> 3;
  for (int i = t; i < n16; i += 256) {
#if HAVE_ASYNC_LDS
    __builtin_amdgcn_global_load_async_to_lds_b128(
        (__attribute__((address_space(1))) v4i*)(src + i * 8),
        (__attribute__((address_space(3))) v4i*)(dst + i * 8), 0, 0);
#else
    *(v8h*)(dst + i * 8) = *(const v8h*)(src + i * 8);
#endif
  }
}

// A fragment (16x32 f16) per documented CDNA5 layout:
// lanes 0-15 : row=lane,    K = kc*32 + {0..7, 16..23}
// lanes 16-31: row=lane-16, K = kc*32 + {8..15, 24..31}
__device__ __forceinline__ v16h lds_a_frag(const _Float16* m, int strideH, int lane, int kc) {
  const _Float16* p = m + (lane & 15) * strideH + kc * 32 + ((lane < 16) ? 0 : 8);
  v8h lo = *(const v8h*)p;
  v8h hi = *(const v8h*)(p + 16);
  v16h a;
#pragma unroll
  for (int i = 0; i < 8; i++) { a[i] = lo[i]; a[i + 8] = hi[i]; }
  return a;
}

// One dense layer: [16 x K] @ [K x 64] + bias -> C in 4 N-chunk accumulators.
// B fragments pre-packed: block (kc*4+nc) of 512 halves, lane-major 16 halves each.
// wp points into LDS (weights staged per block).
__device__ __forceinline__ void mlp_layer(const _Float16* mIn, int strideH, int kcCount,
                                          const _Float16* wp,
                                          const float* __restrict__ bias,
                                          int lane, v8f acc[4]) {
#pragma unroll
  for (int nc = 0; nc < 4; nc++) {
    float b = bias[nc * 16 + (lane & 15)];
#pragma unroll
    for (int j = 0; j < 8; j++) acc[nc][j] = b;
  }
  for (int kc = 0; kc < kcCount; kc++) {
    v16h a = lds_a_frag(mIn, strideH, lane, kc);
#pragma unroll
    for (int nc = 0; nc < 4; nc++) {
      v16h b = *(const v16h*)(wp + ((kc * 4 + nc) << 9) + (lane << 4));
      acc[nc] = __builtin_amdgcn_wmma_f32_16x16x32_f16(
          false, a, false, b, (short)0, acc[nc], false, false);
    }
  }
}

// C layout: lane&15 = column, lanes>=16 hold rows 8..15, VGPR j = row j (+8).
__device__ __forceinline__ void store_c_f16_gelu(_Float16* x, int lane, const v8f acc[4]) {
  int moff = (lane < 16) ? 0 : 8;
  int col0 = lane & 15;
#pragma unroll
  for (int nc = 0; nc < 4; nc++)
#pragma unroll
    for (int j = 0; j < 8; j++)
      x[(moff + j) * 64 + nc * 16 + col0] = (_Float16)gelu_f(acc[nc][j]);
}

__device__ __forceinline__ void store_c_f32(float* x, int lane, const v8f acc[4]) {
  int moff = (lane < 16) ? 0 : 8;
  int col0 = lane & 15;
#pragma unroll
  for (int nc = 0; nc < 4; nc++)
#pragma unroll
    for (int j = 0; j < 8; j++)
      x[(moff + j) * 64 + nc * 16 + col0] = acc[nc][j];
}

// -------------------------------------------------- weight pre-pack (f32 -> f16 B frags)

__global__ void pack_w16(const float* __restrict__ W, int K, int Kpad,
                         _Float16* __restrict__ dst) {
  int idx = blockIdx.x * blockDim.x + threadIdx.x;
  int total = Kpad * 64;
  if (idx >= total) return;
  int blk = idx >> 9;          // (kc*4 + nc)
  int within = idx & 511;
  int lane = within >> 4;      // 0..31
  int e = within & 15;         // element within lane's 16 halves
  int kc = blk >> 2;
  int nc = blk & 3;
  int v = e >> 1;
  int j = e & 1;
  int k = kc * 32 + ((lane < 16) ? 0 : 8) + ((2 * v + j) & 7) + (((2 * v + j) >> 3) << 4);
  int col = nc * 16 + (lane & 15);
  float val = (k < K) ? W[(size_t)k * 64 + col] : 0.0f;
  dst[idx] = (_Float16)val;
}

// -------------------------------------------------- small kernels

__global__ void zero_kernel(float* p, int n) {
  int i = blockIdx.x * blockDim.x + threadIdx.x;
  if (i < n) p[i] = 0.0f;
}

__global__ void encoder_kernel(const float* __restrict__ nodes,
                               const float* __restrict__ W,
                               const float* __restrict__ b,
                               float* __restrict__ h) {
  int idx = blockIdx.x * blockDim.x + threadIdx.x;
  if (idx >= NNODES * DH) return;
  int n = idx >> 6, c = idx & 63;
  float acc = b[c];
#pragma unroll
  for (int k = 0; k < 7; k++) acc += nodes[n * 7 + k] * W[k * 64 + c];
  h[idx] = acc;
}

__global__ void reduce_mean_kernel(const float* __restrict__ h, float* __restrict__ agg) {
  int col = threadIdx.x & 63;
  int grp = blockIdx.x * (blockDim.x >> 6) + (threadIdx.x >> 6);
  int ngrp = gridDim.x * (blockDim.x >> 6);
  float s = 0.0f;
  for (int r = grp; r < NNODES; r += ngrp) s += h[(size_t)r * DH + col];
  atomicAdd(&agg[col], s * (1.0f / (float)NNODES));
}

__global__ void readout_kernel(const float* __restrict__ agg,
                               const float* __restrict__ W1, const float* __restrict__ B1,
                               const float* __restrict__ W2, const float* __restrict__ B2,
                               const float* __restrict__ W3, const float* __restrict__ B3,
                               const float* __restrict__ W4, const float* __restrict__ B4,
                               float* __restrict__ out) {
  __shared__ float x0[64], x1[256], x2[128], x3[128];
  int t = threadIdx.x;
  if (t < 64) x0[t] = agg[t];
  __syncthreads();
  { // 64 -> 256
    float s = B1[t];
    for (int k = 0; k < 64; k++) s += x0[k] * W1[k * 256 + t];
    x1[t] = gelu_f(s);
  }
  __syncthreads();
  if (t < 128) { // 256 -> 128
    float s = B2[t];
    for (int k = 0; k < 256; k++) s += x1[k] * W2[k * 128 + t];
    x2[t] = gelu_f(s);
  }
  __syncthreads();
  if (t < 128) { // 128 -> 128
    float s = B3[t];
    for (int k = 0; k < 128; k++) s += x2[k] * W3[k * 128 + t];
    x3[t] = gelu_f(s);
  }
  __syncthreads();
  if (t == 0) { // 128 -> 1 (no gelu)
    float s = B4[0];
    for (int k = 0; k < 128; k++) s += x3[k] * W4[k];
    out[0] = s;
  }
}

// -------------------------------------------------- edge update (WMMA)

// One wave = one tile of 16 edges. m = [e | h[snd] | h[rcv] | zero-pad] staged f16 in LDS.
// All 8 waves of the block share one LDS copy of the step's weights.
template <int DE, int KPAD, bool EF16>
__global__ __launch_bounds__(256) void edge_kernel(
    const void* __restrict__ ein_v, const float* __restrict__ h,
    const int* __restrict__ senders, const int* __restrict__ receivers,
    const _Float16* __restrict__ w1, const float* __restrict__ b1,
    const _Float16* __restrict__ w2, const float* __restrict__ b2,
    const _Float16* __restrict__ w3, const float* __restrict__ b3,
    const float* __restrict__ lng, const float* __restrict__ lnb,
    _Float16* __restrict__ eout, float* __restrict__ recv) {
  constexpr int REG = (KPAD * 32 > 4096) ? KPAD * 32 : 4096; // bytes per wave tile
  constexpr int W1H = KPAD * 64;                             // halves in layer-1 weights
  __shared__ __align__(16) unsigned char smem[8 * REG];
  __shared__ __align__(32) _Float16 wlds[W1H + 8192];
  __shared__ float stats[8][16][2];
  int wave = threadIdx.x >> 5, lane = threadIdx.x & 31;
  int tile = blockIdx.x * 8 + wave;
  int r0 = tile * 16;
  _Float16* m = (_Float16*)(smem + wave * REG);

  // ---- block-cooperative weight staging (async global->LDS when available)
  {
    int t = threadIdx.x;
    stage_weights(wlds, w1, W1H, t);
    stage_weights(wlds + W1H, w2, 4096, t);
    stage_weights(wlds + W1H + 4096, w3, 4096, t);
    wait_async_all();
    __syncthreads();
  }
  const _Float16* lw1 = wlds;
  const _Float16* lw2 = wlds + W1H;
  const _Float16* lw3 = wlds + W1H + 4096;

  // ---- stage concat row into LDS (2 lanes per row, float4 granules)
  int half = lane >> 4, row = lane & 15;
  int r = r0 + row;
  int snd = senders[r], rcv = receivers[r];
  const float* hs = h + (size_t)snd * DH;
  const float* hr = h + (size_t)rcv * DH;
  const int KH = KPAD / 2;
  for (int c = half * KH; c < half * KH + KH; c += 4) {
    float4 v;
    if (c < DE) {
      if (EF16) {
        h4 ev = *(const h4*)((const _Float16*)ein_v + (size_t)r * DE + c);
        v.x = (float)ev[0]; v.y = (float)ev[1]; v.z = (float)ev[2]; v.w = (float)ev[3];
      } else {
        v = *(const float4*)((const float*)ein_v + (size_t)r * DE + c);
      }
    } else if (c < DE + DH) {
      v = *(const float4*)(hs + (c - DE));
    } else if (c < DE + 2 * DH) {
      v = *(const float4*)(hr + (c - DE - DH));
    } else {
      v = make_float4(0.f, 0.f, 0.f, 0.f);
    }
    h4 p;
    p[0] = (_Float16)v.x; p[1] = (_Float16)v.y; p[2] = (_Float16)v.z; p[3] = (_Float16)v.w;
    *(h4*)(m + row * KPAD + c) = p;
  }
  wave_lds_sync();

  // ---- 3-layer MLP via WMMA
  v8f acc[4];
  mlp_layer(m, KPAD, KPAD / 32, lw1, b1, lane, acc);
  wave_lds_sync();
  store_c_f16_gelu(m, lane, acc);
  wave_lds_sync();
  mlp_layer(m, 64, 2, lw2, b2, lane, acc);
  wave_lds_sync();
  store_c_f16_gelu(m, lane, acc);
  wave_lds_sync();
  mlp_layer(m, 64, 2, lw3, b3, lane, acc);
  wave_lds_sync();

  float* xf = (float*)(smem + wave * REG);
  store_c_f32(xf, lane, acc);
  wave_lds_sync();

  // ---- segment_sum(e_new, receivers): coalesced f32 atomics
  for (int M = 0; M < 16; M++) {
    int rr = receivers[r0 + M];
    atomicAdd(&recv[(size_t)rr * DH + lane], xf[M * 64 + lane]);
    atomicAdd(&recv[(size_t)rr * DH + 32 + lane], xf[M * 64 + 32 + lane]);
  }

  // ---- LayerNorm per edge row, write f16 edge state
  if (lane < 16) {
    float mu = 0.f;
    for (int c = 0; c < 64; c++) mu += xf[lane * 64 + c];
    mu *= (1.f / 64.f);
    float var = 0.f;
    for (int c = 0; c < 64; c++) { float d = xf[lane * 64 + c] - mu; var += d * d; }
    var *= (1.f / 64.f);
    stats[wave][lane][0] = mu;
    stats[wave][lane][1] = rsqrtf(var + 1e-6f);
  }
  wave_lds_sync();
  float g0 = lng[lane], g1 = lng[lane + 32], be0 = lnb[lane], be1 = lnb[lane + 32];
  for (int M = 0; M < 16; M++) {
    float mu = stats[wave][M][0], rs = stats[wave][M][1];
    size_t o = (size_t)(r0 + M) * DH;
    eout[o + lane]      = (_Float16)((xf[M * 64 + lane] - mu) * rs * g0 + be0);
    eout[o + 32 + lane] = (_Float16)((xf[M * 64 + 32 + lane] - mu) * rs * g1 + be1);
  }
}

// -------------------------------------------------- node update (WMMA)

__global__ __launch_bounds__(256) void node_kernel(
    float* __restrict__ h, const float* __restrict__ recv,
    const _Float16* __restrict__ w1, const float* __restrict__ b1,
    const _Float16* __restrict__ w2, const float* __restrict__ b2,
    const _Float16* __restrict__ w3, const float* __restrict__ b3,
    const float* __restrict__ lng, const float* __restrict__ lnb) {
  constexpr int KPAD = 128;
  constexpr int REG = 4096; // == KPAD*16*2 == 16*64*4
  constexpr int W1H = KPAD * 64;
  __shared__ __align__(16) unsigned char smem[8 * REG];
  __shared__ __align__(32) _Float16 wlds[W1H + 8192];
  __shared__ float stats[8][16][2];
  int wave = threadIdx.x >> 5, lane = threadIdx.x & 31;
  int tile = blockIdx.x * 8 + wave;
  bool active = (tile < NNODES / 16);

  { // all waves stage weights, then one block-wide barrier
    int t = threadIdx.x;
    stage_weights(wlds, w1, W1H, t);
    stage_weights(wlds + W1H, w2, 4096, t);
    stage_weights(wlds + W1H + 4096, w3, 4096, t);
    wait_async_all();
    __syncthreads();
  }
  if (!active) return;
  const _Float16* lw1 = wlds;
  const _Float16* lw2 = wlds + W1H;
  const _Float16* lw3 = wlds + W1H + 4096;

  int r0 = tile * 16;
  _Float16* m = (_Float16*)(smem + wave * REG);

  int half = lane >> 4, row = lane & 15;
  int r = r0 + row;
  const float* s0 = h + (size_t)r * DH;
  const float* s1 = recv + (size_t)r * DH;
  for (int c = half * 64; c < half * 64 + 64; c += 4) {
    float4 v = (c < 64) ? *(const float4*)(s0 + c) : *(const float4*)(s1 + c - 64);
    h4 p;
    p[0] = (_Float16)v.x; p[1] = (_Float16)v.y; p[2] = (_Float16)v.z; p[3] = (_Float16)v.w;
    *(h4*)(m + row * KPAD + c) = p;
  }
  wave_lds_sync();

  v8f acc[4];
  mlp_layer(m, KPAD, 4, lw1, b1, lane, acc);
  wave_lds_sync();
  store_c_f16_gelu(m, lane, acc);
  wave_lds_sync();
  mlp_layer(m, 64, 2, lw2, b2, lane, acc);
  wave_lds_sync();
  store_c_f16_gelu(m, lane, acc);
  wave_lds_sync();
  mlp_layer(m, 64, 2, lw3, b3, lane, acc);
  wave_lds_sync();

  float* xf = (float*)(smem + wave * REG);
  store_c_f32(xf, lane, acc);
  wave_lds_sync();

  if (lane < 16) {
    float mu = 0.f;
    for (int c = 0; c < 64; c++) mu += xf[lane * 64 + c];
    mu *= (1.f / 64.f);
    float var = 0.f;
    for (int c = 0; c < 64; c++) { float d = xf[lane * 64 + c] - mu; var += d * d; }
    var *= (1.f / 64.f);
    stats[wave][lane][0] = mu;
    stats[wave][lane][1] = rsqrtf(var + 1e-6f);
  }
  wave_lds_sync();
  float g0 = lng[lane], g1 = lng[lane + 32], be0 = lnb[lane], be1 = lnb[lane + 32];
  for (int M = 0; M < 16; M++) {
    float mu = stats[wave][M][0], rs = stats[wave][M][1];
    size_t o = (size_t)(r0 + M) * DH;
    h[o + lane]      = (xf[M * 64 + lane] - mu) * rs * g0 + be0;
    h[o + 32 + lane] = (xf[M * 64 + 32 + lane] - mu) * rs * g1 + be1;
  }
}

// -------------------------------------------------- launch

extern "C" void kernel_launch(void* const* d_in, const int* in_sizes, int n_in,
                              void* d_out, int out_size, void* d_ws, size_t ws_size,
                              hipStream_t stream) {
  (void)in_sizes; (void)n_in; (void)out_size; (void)ws_size;
  const float* nodes     = (const float*)d_in[0];
  const float* edges     = (const float*)d_in[1];
  const int*   senders   = (const int*)d_in[2];
  const int*   receivers = (const int*)d_in[3];
  const float* encW      = (const float*)d_in[4];
  const float* encB      = (const float*)d_in[5];
  auto F = [&](int i) { return (const float*)d_in[i]; };
  // leaf order: edge_params @6 (3 steps x 3 layers x (W,b)), node_params @24,
  // ln_node @42, ln_edge @48, ro_params @54

  char* ws = (char*)d_ws;
  size_t off = 0;
  auto alloc = [&](size_t bytes) -> void* {
    void* p = ws + off;
    off += (bytes + 255) & ~(size_t)255;
    return p;
  };
  float*    h    = (float*)alloc((size_t)NNODES * DH * 4);
  _Float16* e    = (_Float16*)alloc((size_t)NEDGES * DH * 2);
  float*    recv = (float*)alloc((size_t)NNODES * DH * 4);
  float*    agg  = (float*)alloc(64 * 4);

  const int eK[3]  = {144, 192, 192};
  const int eKp[3] = {160, 192, 192};
  _Float16* ewp[3][3];
  _Float16* nwp[3][3];
  for (int s = 0; s < 3; s++)
    for (int l = 0; l < 3; l++) {
      int Kp = (l == 0) ? eKp[s] : 64;
      ewp[s][l] = (_Float16*)alloc((size_t)Kp * 64 * 2);
      int Kn = (l == 0) ? 128 : 64;
      nwp[s][l] = (_Float16*)alloc((size_t)Kn * 64 * 2);
    }

  // ---- pack weights to f16 B-fragment layout
  for (int s = 0; s < 3; s++)
    for (int l = 0; l < 3; l++) {
      const float* We = F(6 + s * 6 + l * 2);
      int K = (l == 0) ? eK[s] : 64, Kp = (l == 0) ? eKp[s] : 64;
      pack_w16<<<(Kp * 64 + 255) / 256, 256, 0, stream>>>(We, K, Kp, ewp[s][l]);
      const float* Wn = F(24 + s * 6 + l * 2);
      int Kn = (l == 0) ? 128 : 64;
      pack_w16<<<(Kn * 64 + 255) / 256, 256, 0, stream>>>(Wn, Kn, Kn, nwp[s][l]);
    }

  // ---- encoder
  encoder_kernel<<<(NNODES * DH + 255) / 256, 256, 0, stream>>>(nodes, encW, encB, h);

  // ---- message-passing steps
  for (int s = 0; s < 3; s++) {
    zero_kernel<<<(NNODES * DH + 255) / 256, 256, 0, stream>>>(recv, NNODES * DH);
    const float* b1 = F(6 + s * 6 + 1);
    const float* b2 = F(6 + s * 6 + 3);
    const float* b3 = F(6 + s * 6 + 5);
    const float* lnEg = F(48 + s * 2), * lnEb = F(49 + s * 2);
    const float* lnNg = F(42 + s * 2), * lnNb = F(43 + s * 2);
    if (s == 0) {
      edge_kernel<16, 160, false><<<NEDGES / 128, 256, 0, stream>>>(
          (const void*)edges, h, senders, receivers,
          ewp[0][0], b1, ewp[0][1], b2, ewp[0][2], b3, lnEg, lnEb, e, recv);
    } else {
      edge_kernel<64, 192, true><<<NEDGES / 128, 256, 0, stream>>>(
          (const void*)e, h, senders, receivers,
          ewp[s][0], b1, ewp[s][1], b2, ewp[s][2], b3, lnEg, lnEb, e, recv);
    }
    const float* nb1 = F(24 + s * 6 + 1);
    const float* nb2 = F(24 + s * 6 + 3);
    const float* nb3 = F(24 + s * 6 + 5);
    node_kernel<<<(NNODES / 16 + 7) / 8, 256, 0, stream>>>(
        h, recv, nwp[s][0], nb1, nwp[s][1], nb2, nwp[s][2], nb3, lnNg, lnNb);
  }

  // ---- readout
  zero_kernel<<<1, 64, 0, stream>>>(agg, 64);
  reduce_mean_kernel<<<256, 256, 0, stream>>>(h, agg);
  readout_kernel<<<1, 256, 0, stream>>>(agg, F(54), F(55), F(56), F(57),
                                        F(58), F(59), F(60), F(61), (float*)d_out);
}
